// GraphConvModel_77146202571216
// MI455X (gfx1250) — matrix operand order repaired and checked by reading, not compiled, not run on hardware
//
#include <hip/hip_runtime.h>

// ---------------------------------------------------------------------------
// MI455X (gfx1250) implementation of the GCN+LSTM+deconv policy network.
// ~160 of ~167 GFLOP are GEMMs -> WMMA pipe, f16 in / f32 accumulate.
// GEMM: 8 waves/block, each wave owns a 32x32 output block (2x2 WMMA tiles),
// macro tile 64x128, K staged in 32-chunks through LDS via ASYNC global->LDS
// b128 copies (both A and B tiles are straight copies: every B operand is
// stored K-contiguous ("[N][K]") -- weights pre-transposed once, the per-step
// activation B (tmp) written blocked-transposed by the producing GEMM).
// ---------------------------------------------------------------------------

typedef __attribute__((ext_vector_type(16))) _Float16 v16h;
typedef __attribute__((ext_vector_type(8)))  _Float16 v8h;
typedef __attribute__((ext_vector_type(8)))  float    v8f;

#define TM 64
#define TN 128
#define TK 32

// async global -> LDS copy of 16 bytes (per active lane), ASYNCcnt tracked
__device__ __forceinline__ void async_b128(const _Float16* gsrc, _Float16* ldst) {
    unsigned lds_off = (unsigned)(unsigned long long)(void*)ldst;  // LDS aperture: low 32 bits
    asm volatile("global_load_async_to_lds_b128 %0, %1, off"
                 :: "v"(lds_off), "v"((unsigned long long)gsrc)
                 : "memory");
}
__device__ __forceinline__ void wait_async() {
    asm volatile("s_wait_asynccnt 0" ::: "memory");
}

// ---------------------------------------------------------------------------
// C[M,N] = act( A[M,K](f16 rm, lda) * B^T[N,K](f16 rm, ldb) + bias )
// Requires M%64==0, N%128==0, K%32==0 (all call sites padded accordingly).
// act: 0=none, 1=tanh, 2=leaky_relu(0.1)
// c16mode: 0 -> C16[row*ldc16+col]; 1 -> blocked transpose (block 128):
//          C16[((row>>7)*ldc16 + col)*128 + (row&127)]
// ---------------------------------------------------------------------------
__global__ __launch_bounds__(256) void k_gemm(
    const _Float16* __restrict__ A, const _Float16* __restrict__ B,
    const float* __restrict__ bias, float* __restrict__ C32,
    _Float16* __restrict__ C16,
    int M, int N, int K, int lda, int ldb, int ldc32, int ldc16, int c16mode,
    long sA, long sB, long sC32, long sC16, int act)
{
    __shared__ __align__(16) _Float16 sAt[TM][TK];   // 64x32
    __shared__ __align__(16) _Float16 sBt[TN][TK];   // 128x32, [n][k]

    const long z = blockIdx.z;
    A += z * sA;
    B += z * sB;
    if (C32) C32 += z * sC32;
    if (C16) C16 += z * sC16;

    const int tid  = threadIdx.x;
    const int lane = tid & 31;
    const int wave = tid >> 5;       // 0..7
    const int wm   = wave >> 2;      // 0..1 -> 32-row slab
    const int wn   = wave & 3;       // 0..3 -> 32-col slab
    const int m0   = blockIdx.y * TM;
    const int n0   = blockIdx.x * TN;

    const int hl = lane >> 4;        // lane half
    const int ll = lane & 15;

    // async-copy addressing (per thread)
    const int ar = tid >> 2,  ac = (tid & 3) << 3;   // A: 64 rows x (4x8) cols
    const int br = tid >> 1,  bc = (tid & 1) << 4;   // B: 128 rows x (2x16) cols

    v8f acc[2][2] = {};

    for (int k0 = 0; k0 < K; k0 += TK) {
        __syncthreads();             // previous iteration's fragment reads done
        // A tile: one b128 per thread
        async_b128(A + (long)(m0 + ar) * lda + k0 + ac, &sAt[ar][ac]);
        // B tile: two b128 per thread
        {
            const _Float16* src = B + (long)(n0 + br) * ldb + k0 + bc;
            async_b128(src,     &sBt[br][bc]);
            async_b128(src + 8, &sBt[br][bc + 8]);
        }
        wait_async();
        __syncthreads();

        // fragment packing per CDNA5 ISA 7.12.2 (16-bit A 16x32 / B 32x16)
        v16h af[2], bf[2];
        #pragma unroll
        for (int i = 0; i < 2; ++i) {
            const _Float16* ap = &sAt[wm * 32 + i * 16 + ll][0];
            #pragma unroll
            for (int j = 0; j < 8; ++j) af[i][j]     = ap[hl * 8 + j];
            #pragma unroll
            for (int j = 0; j < 8; ++j) af[i][8 + j] = ap[16 + hl * 8 + j];
        }
        #pragma unroll
        for (int i = 0; i < 2; ++i) {
            const _Float16* bp = &sBt[wn * 32 + i * 16 + ll][0];
            #pragma unroll
            for (int j = 0; j < 8; ++j) bf[i][j]     = bp[hl * 8 + j];
            #pragma unroll
            for (int j = 0; j < 8; ++j) bf[i][8 + j] = bp[16 + hl * 8 + j];
        }
        #pragma unroll
        for (int i = 0; i < 2; ++i)
            #pragma unroll
            for (int j = 0; j < 2; ++j)
                acc[i][j] = __builtin_amdgcn_wmma_f32_16x16x32_f16(
                    false, af[i], false, bf[j], (short)0, acc[i][j], false, false);
    }

    // epilogue: lane -> col = lane%16, VGPR r -> row = (lane/16)*8 + r
    #pragma unroll
    for (int i = 0; i < 2; ++i) {
        #pragma unroll
        for (int j = 0; j < 2; ++j) {
            const int col  = n0 + wn * 32 + j * 16 + ll;
            const int row0 = m0 + wm * 32 + i * 16 + hl * 8;
            const float bb = bias ? bias[col] : 0.f;
            #pragma unroll
            for (int r = 0; r < 8; ++r) {
                float v = acc[i][j][r] + bb;
                if (act == 1)      v = tanhf(v);
                else if (act == 2) v = v > 0.f ? v : 0.1f * v;
                const long row = row0 + r;
                if (C32) C32[row * (long)ldc32 + col] = v;
                if (C16) {
                    long idx = (c16mode == 0)
                        ? row * (long)ldc16 + col
                        : ((row >> 7) * (long)ldc16 + col) * 128 + (row & 127);
                    C16[idx] = (_Float16)v;
                }
            }
        }
    }
}

// --------------------------- small prep kernels -----------------------------
// d[n*Kpad + k] = (k < Ksrc) ? (f16)s[k*N + n] : 0    (B^T with K padding)
__global__ void k_cvtT(const float* __restrict__ s, _Float16* __restrict__ d,
                       int N, int Ksrc, int Kpad) {
    long i = (long)blockIdx.x * 256 + threadIdx.x;
    if (i >= (long)N * Kpad) return;
    int n = (int)(i / Kpad), k = (int)(i % Kpad);
    d[i] = (k < Ksrc) ? (_Float16)s[(long)k * N + n] : (_Float16)0.f;
}
// Wih/Whh [1024,256] -> Wcat^T [1024][512] (rows already K-major)
__global__ void k_cvt_cat(const float* __restrict__ s, _Float16* __restrict__ d, int off) {
    int i = blockIdx.x * 256 + threadIdx.x;
    if (i < 1024 * 256) {
        int n = i >> 8, k = i & 255;
        d[n * 512 + off + k] = (_Float16)s[i];
    }
}
__global__ void k_bsum(const float* a, const float* b, float* d, int n) {
    int i = blockIdx.x * 256 + threadIdx.x;
    if (i < n) d[i] = a[i] + b[i];
}
__global__ void k_bt1e(const float* b, float* d) {        // expand per-channel bias
    int i = blockIdx.x * 256 + threadIdx.x;
    if (i < 4096) d[i] = b[i >> 4];
}
__global__ void k_init(const float* __restrict__ LSTMh, float* __restrict__ h,
                       float* __restrict__ c, _Float16* __restrict__ cat) {
    int i = blockIdx.x * 256 + threadIdx.x;
    if (i < 16384) {
        float hv = LSTMh[i];
        h[i] = hv;
        c[i] = LSTMh[16384 + i];
        int b = i >> 8, j = i & 255;
        cat[b * 512 + 256 + j] = (_Float16)hv;
    }
}

// --------------------------- per-step kernels -------------------------------
__global__ void k_adj(const float* __restrict__ G, _Float16* __restrict__ A16, int t) {
    int b = blockIdx.x;                 // 64
    int tid = threadIdx.x;              // 128
    __shared__ float dis[128];
    const float* Gb = G + ((long)(b * 32 + t)) * 128 * 128;
    float s = 1.0f;                     // +I diagonal
    for (int j = 0; j < 128; ++j) s += Gb[tid * 128 + j];
    dis[tid] = rsqrtf(fmaxf(s, 1.0f));
    __syncthreads();
    _Float16* Ab = A16 + (long)b * 128 * 128;
    for (int i = 0; i < 128; ++i) {
        float a = Gb[i * 128 + tid] + (i == tid ? 1.f : 0.f);
        Ab[i * 128 + tid] = (_Float16)(a * dis[i] * dis[tid]);
    }
}
__global__ void k_gatherx(const float* __restrict__ X, _Float16* __restrict__ Xg, int t) {
    long i = (long)blockIdx.x * 256 + threadIdx.x;
    if (i >= 524288) return;
    long bn = i >> 6, u = i & 63;
    long b = bn >> 7, node = bn & 127;
    Xg[i] = (_Float16)X[(((b * 32) + t) * 128 + node) * 64 + u];
}
__global__ void k_goa(const _Float16* __restrict__ h3, const float* __restrict__ pa,
                      float* __restrict__ goa, _Float16* __restrict__ goah, int t) {
    int b = blockIdx.x;                 // 64
    int j = threadIdx.x;                // 288
    if (j < 256) {
        float m = -1e30f;
        const _Float16* p = h3 + (long)b * 128 * 256 + j;
        for (int node = 0; node < 128; ++node) m = fmaxf(m, (float)p[node * 256]);
        goa[b * 268 + j] = m;
        goah[b * 288 + j] = (_Float16)m;
    } else if (j < 268) {
        float v = pa[(b * 32 + t) * 12 + (j - 256)];
        goa[b * 268 + j] = v;
        goah[b * 288 + j] = (_Float16)v;
    } else {
        goah[b * 288 + j] = (_Float16)0.f;
    }
}
__device__ __forceinline__ float sigf(float x) { return 1.f / (1.f + __expf(-x)); }
__global__ void k_lstm(const float* __restrict__ gates, float* __restrict__ h,
                       float* __restrict__ c, float* __restrict__ hout,
                       _Float16* __restrict__ cat, const int* __restrict__ rel, int t) {
    int b = blockIdx.x, j = threadIdx.x;          // 64 x 256
    const float* g = gates + b * 1024;
    float i_ = sigf(g[j]);
    float f_ = sigf(g[256 + j]);
    float gg = tanhf(g[512 + j]);
    float o_ = sigf(g[768 + j]);
    float c_new = f_ * c[b * 256 + j] + i_ * gg;
    float h_new = o_ * tanhf(c_new);
    float keep = (rel[b * 32 + t] != 0) ? 1.f : 0.f;
    hout[b * 256 + j] = h_new;
    h[b * 256 + j] = h_new * keep;
    c[b * 256 + j] = c_new * keep;
    cat[b * 512 + 256 + j] = (_Float16)(h_new * keep);
}

// --------------------------- tail kernels -----------------------------------
__global__ void k_glo(const float* __restrict__ hout, const float* __restrict__ goa,
                      float* __restrict__ glo, _Float16* __restrict__ gloh) {
    int i = blockIdx.x * 256 + threadIdx.x;
    if (i >= 64 * 544) return;
    int b = i / 544, j = i % 544;
    float v = 0.f;
    if (j < 256)      v = hout[b * 256 + j];
    else if (j < 524) v = goa[b * 268 + j - 256];
    gloh[b * 544 + j] = (_Float16)v;
    if (j < 524) glo[b * 524 + j] = v;
}
__global__ void k_heads(const float* __restrict__ glo, const float* __restrict__ Wa,
                        const float* __restrict__ ba, const float* __restrict__ Wv,
                        const float* __restrict__ bv, const int* __restrict__ avail,
                        float* __restrict__ out_ns, float* __restrict__ out_val) {
    int b = threadIdx.x;
    if (b >= 64) return;
    const float* g = glo + b * 524;
    float lg[8];
    for (int a = 0; a < 8; ++a) {
        float s = ba[a];
        for (int c = 0; c < 524; ++c) s += g[c] * Wa[c * 8 + a];
        lg[a] = s;
    }
    const float NEG = -__builtin_inff();
    for (int a = 0; a < 8; ++a) if (avail[b * 8 + a] == 0) lg[a] = NEG;
    float m = NEG;
    for (int a = 0; a < 8; ++a) m = fmaxf(m, lg[a]);
    float s = 0.f;
    for (int a = 0; a < 8; ++a) s += __expf(lg[a] - m);
    for (int a = 0; a < 8; ++a) out_ns[b * 8 + a] = __expf(lg[a] - m) / s;
    float v = bv[0];
    for (int c = 0; c < 524; ++c) v += g[c] * Wv[c];
    out_val[b] = v;
}
__global__ void k_conv3(const float* __restrict__ src, const float* __restrict__ w,
                        const float* __restrict__ bias, float* __restrict__ dst,
                        int Nb, int Cin, int Cout, int H, int W) {
    long idx = (long)blockIdx.x * 256 + threadIdx.x;
    long total = (long)Nb * Cout * H * W;
    if (idx >= total) return;
    int x = idx % W;
    int y = (idx / W) % H;
    int co = (int)((idx / ((long)W * H)) % Cout);
    int n  = (int)(idx / ((long)W * H * Cout));
    float acc = bias[co];
    for (int ci = 0; ci < Cin; ++ci) {
        const float* ch = src + ((long)(n * Cin + ci)) * H * W;
        const float* wc = w + ((long)co * Cin + ci) * 9;
        #pragma unroll
        for (int ky = 0; ky < 3; ++ky) {
            int iy = y + ky - 1;
            if ((unsigned)iy >= (unsigned)H) continue;
            #pragma unroll
            for (int kx = 0; kx < 3; ++kx) {
                int ix = x + kx - 1;
                if ((unsigned)ix >= (unsigned)W) continue;
                acc += ch[iy * W + ix] * wc[ky * 3 + kx];
            }
        }
    }
    dst[idx] = acc;
}
__device__ __forceinline__ float up2_relu(const float* __restrict__ ch,
                                          int Hs, int Ws, int v, int u) {
    float sv = 0.5f * v - 0.25f, su = 0.5f * u - 0.25f;
    int i0 = (int)floorf(sv), j0 = (int)floorf(su);
    float fy = sv - i0, fx = su - j0;
    int i1 = i0 + 1 < Hs - 1 ? i0 + 1 : Hs - 1;
    int j1 = j0 + 1 < Ws - 1 ? j0 + 1 : Ws - 1;
    i0 = i0 > 0 ? i0 : 0;
    j0 = j0 > 0 ? j0 : 0;
    float a = ch[i0 * Ws + j0], b = ch[i0 * Ws + j1];
    float c = ch[i1 * Ws + j0], d = ch[i1 * Ws + j1];
    float val = (1.f - fy) * ((1.f - fx) * a + fx * b) +
                fy * ((1.f - fx) * c + fx * d);
    return fmaxf(val, 0.f);
}
__global__ void k_conv3_up(const float* __restrict__ src, const float* __restrict__ w,
                           const float* __restrict__ bias, float* __restrict__ dst,
                           int Nb, int Cin, int Cout, int Hs, int Ws) {
    int Hu = Hs * 2, Wu = Ws * 2;
    long idx = (long)blockIdx.x * 256 + threadIdx.x;
    long total = (long)Nb * Cout * Hu * Wu;
    if (idx >= total) return;
    int x = idx % Wu;
    int y = (idx / Wu) % Hu;
    int co = (int)((idx / ((long)Wu * Hu)) % Cout);
    int n  = (int)(idx / ((long)Wu * Hu * Cout));
    float acc = bias[co];
    for (int ci = 0; ci < Cin; ++ci) {
        const float* ch = src + ((long)(n * Cin + ci)) * Hs * Ws;
        const float* wc = w + ((long)co * Cin + ci) * 9;
        #pragma unroll
        for (int ky = 0; ky < 3; ++ky) {
            int v = y + ky - 1;
            if ((unsigned)v >= (unsigned)Hu) continue;
            #pragma unroll
            for (int kx = 0; kx < 3; ++kx) {
                int u = x + kx - 1;
                if ((unsigned)u >= (unsigned)Wu) continue;
                acc += up2_relu(ch, Hs, Ws, v, u) * wc[ky * 3 + kx];
            }
        }
    }
    dst[idx] = acc;
}
__global__ void k_softmax_sp(const float* __restrict__ src, float* __restrict__ dst) {
    int bc = blockIdx.x;                                  // 64*2
    const float* p = src + (long)bc * 4096;
    float* q = dst + (long)bc * 4096;
    __shared__ float red[256];
    int tid = threadIdx.x;
    float m = -1e30f;
    for (int i = tid; i < 4096; i += 256) m = fmaxf(m, p[i]);
    red[tid] = m; __syncthreads();
    for (int s = 128; s > 0; s >>= 1) {
        if (tid < s) red[tid] = fmaxf(red[tid], red[tid + s]);
        __syncthreads();
    }
    m = red[0]; __syncthreads();
    float sum = 0.f;
    for (int i = tid; i < 4096; i += 256) sum += __expf(p[i] - m);
    red[tid] = sum; __syncthreads();
    for (int s = 128; s > 0; s >>= 1) {
        if (tid < s) red[tid] += red[tid + s];
        __syncthreads();
    }
    float inv = 1.0f / red[0];
    for (int i = tid; i < 4096; i += 256) q[i] = __expf(p[i] - m) * inv;
}
__global__ void k_outlstm(const float* __restrict__ h, const float* __restrict__ c,
                          float* __restrict__ dst) {
    int i = blockIdx.x * 256 + threadIdx.x;
    if (i < 32768) dst[i] = (i < 16384) ? h[i] : c[i - 16384];
}

// ---------------------------------------------------------------------------
extern "C" void kernel_launch(void* const* d_in, const int* in_sizes, int n_in,
                              void* d_out, int out_size, void* d_ws, size_t ws_size,
                              hipStream_t stream)
{
    (void)in_sizes; (void)n_in; (void)out_size; (void)ws_size;
    const float* G    = (const float*)d_in[0];
    const float* X    = (const float*)d_in[1];
    const int*   avail= (const int*)  d_in[2];
    const float* LSTMh= (const float*)d_in[3];
    const float* pact = (const float*)d_in[4];
    const int*   rel  = (const int*)  d_in[5];
    const float* We  = (const float*)d_in[6];  const float* be  = (const float*)d_in[7];
    const float* W1  = (const float*)d_in[8];  const float* b1  = (const float*)d_in[9];
    const float* W2  = (const float*)d_in[10]; const float* b2  = (const float*)d_in[11];
    const float* W3  = (const float*)d_in[12]; const float* b3  = (const float*)d_in[13];
    const float* Wle = (const float*)d_in[14]; const float* ble = (const float*)d_in[15];
    const float* Wih = (const float*)d_in[16]; const float* Whh = (const float*)d_in[17];
    const float* bih = (const float*)d_in[18]; const float* bhh = (const float*)d_in[19];
    const float* Wa  = (const float*)d_in[20]; const float* ba  = (const float*)d_in[21];
    const float* Wv  = (const float*)d_in[22]; const float* bv  = (const float*)d_in[23];
    const float* Wt1 = (const float*)d_in[24]; const float* bt1 = (const float*)d_in[25];
    const float* Wc2 = (const float*)d_in[26]; const float* bc2 = (const float*)d_in[27];
    const float* Wc3 = (const float*)d_in[28]; const float* bc3 = (const float*)d_in[29];
    const float* Wc4 = (const float*)d_in[30]; const float* bc4 = (const float*)d_in[31];
    const float* Wc5 = (const float*)d_in[32]; const float* bc5 = (const float*)d_in[33];
    const float* Wc6 = (const float*)d_in[34]; const float* bc6 = (const float*)d_in[35];
    float* out = (float*)d_out;

    // workspace arena (256B-aligned bump allocator)
    char* base = (char*)d_ws;
    size_t off = 0;
    auto allocB = [&](size_t bytes) -> void* {
        void* p = base + off;
        off = (off + bytes + 255) & ~(size_t)255;
        return p;
    };
    auto allocH = [&](size_t n) { return (_Float16*)allocB(n * 2); };
    auto allocF = [&](size_t n) { return (float*)allocB(n * 4); };

    _Float16* hWeT  = allocH(256 * 64);        // [N=256][K=64]
    _Float16* hW1T  = allocH(256 * 256);
    _Float16* hW2T  = allocH(256 * 256);
    _Float16* hW3T  = allocH(256 * 256);
    _Float16* hWleT = allocH(256 * 288);       // K padded 268 -> 288
    _Float16* hWcat = allocH(1024 * 512);      // [N=1024][K=512] = [Wih | Whh] rows
    _Float16* hWt1T = allocH(4096 * 544);      // K padded 524 -> 544
    float*    fBsum = allocF(1024);
    float*    fBt1e = allocF(4096);
    _Float16* hXg   = allocH(8192 * 64);
    _Float16* hA    = allocH(64 * 128 * 128);
    _Float16* hActA = allocH(8192 * 256);      // row-major activations (A operand)
    _Float16* hActBT= allocH(8192 * 256);      // blocked-transposed tmp (B operand)
    float*    fGoa  = allocF(64 * 268);
    _Float16* hGoa  = allocH(64 * 288);
    _Float16* hCat  = allocH(64 * 512);
    float*    fGates= allocF(64 * 1024);
    float*    fH    = allocF(64 * 256);
    float*    fC    = allocF(64 * 256);
    float*    fHout = allocF(64 * 256);
    float*    fGlo  = allocF(64 * 524);
    _Float16* hGlo  = allocH(64 * 544);
    float*    fS1   = allocF(64 * 4096);
    float*    fC2   = allocF(64 * 128 * 16);
    float*    fC3   = allocF(64 * 64 * 64);
    float*    fC4   = allocF(64 * 64 * 256);
    float*    fC5   = allocF(64 * 64 * 1024);
    float*    fC6   = allocF(64 * 2 * 4096);

    auto gb = [](long n) { return dim3((unsigned)((n + 255) / 256)); };
    auto gemm = [&](const _Float16* A, const _Float16* B, const float* bias,
                    float* C32, _Float16* C16, int M, int N, int K,
                    int lda, int ldb, int ldc32, int ldc16, int c16mode,
                    long sA, long sB, long sC32, long sC16, int act, int batches) {
        dim3 grid(N / TN, M / TM, batches);
        k_gemm<<<grid, 256, 0, stream>>>(A, B, bias, C32, C16, M, N, K,
                                         lda, ldb, ldc32, ldc16, c16mode,
                                         sA, sB, sC32, sC16, act);
    };

    // ---- one-time weight preparation (B operands stored [N][Kpad]) --------
    k_cvtT   <<<gb(256 * 64),    256, 0, stream>>>(We,  hWeT,  256, 64,  64);
    k_cvtT   <<<gb(256 * 256),   256, 0, stream>>>(W1,  hW1T,  256, 256, 256);
    k_cvtT   <<<gb(256 * 256),   256, 0, stream>>>(W2,  hW2T,  256, 256, 256);
    k_cvtT   <<<gb(256 * 256),   256, 0, stream>>>(W3,  hW3T,  256, 256, 256);
    k_cvtT   <<<gb(256 * 288),   256, 0, stream>>>(Wle, hWleT, 256, 268, 288);
    k_cvt_cat<<<gb(1024 * 256),  256, 0, stream>>>(Wih, hWcat, 0);
    k_cvt_cat<<<gb(1024 * 256),  256, 0, stream>>>(Whh, hWcat, 256);
    k_bsum   <<<gb(1024),        256, 0, stream>>>(bih, bhh, fBsum, 1024);
    k_cvtT   <<<gb(4096 * 544),  256, 0, stream>>>(Wt1, hWt1T, 4096, 524, 544);
    k_bt1e   <<<gb(4096),        256, 0, stream>>>(bt1, fBt1e);
    k_init   <<<gb(16384),       256, 0, stream>>>(LSTMh, fH, fC, hCat);

    // ---- recurrent scan over 32 timesteps --------------------------------
    const _Float16* WlT[3] = {hW1T, hW2T, hW3T};
    const float*    bl[3]  = {b1, b2, b3};
    for (int t = 0; t < 32; ++t) {
        k_adj    <<<64, 128, 0, stream>>>(G, hA, t);
        k_gatherx<<<gb(524288), 256, 0, stream>>>(X, hXg, t);
        // emb = tanh(Xc @ We + be)        [8192,64] x [64,256]
        gemm(hXg, hWeT, be, nullptr, hActA, 8192, 256, 64,
             64, 64, 0, 256, 0, 0, 0, 0, 0, 1, 1);
        for (int l = 0; l < 3; ++l) {
            // tmp = H @ Wl + bl  -> blocked-transposed C16 ([b][256][128])
            gemm(hActA, WlT[l], bl[l], nullptr, hActBT, 8192, 256, 256,
                 256, 256, 0, 256, 1, 0, 0, 0, 0, 0, 1);
            // H = tanh(A_agg @ tmp)  batched: [128,128] x tmpT[256][128] x 64
            gemm(hA, hActBT, nullptr, nullptr, hActA, 128, 256, 128,
                 128, 128, 0, 256, 0, 128 * 128, 128 * 256, 0, 128 * 256, 1, 64);
        }
        k_goa<<<64, 288, 0, stream>>>(hActA, pact, fGoa, hGoa, t);
        // xin = tanh(goa @ Wle + ble) -> cat[:, 0:256]   [64,288] x [288,256]
        gemm(hGoa, hWleT, ble, nullptr, hCat, 64, 256, 288,
             288, 288, 0, 512, 0, 0, 0, 0, 0, 1, 1);
        // gates = [xin|h] @ [Wih^T;Whh^T] + (bih+bhh)    [64,512] x [512,1024]
        gemm(hCat, hWcat, fBsum, fGates, nullptr, 64, 1024, 512,
             512, 512, 1024, 0, 0, 0, 0, 0, 0, 0, 1);
        k_lstm<<<64, 256, 0, stream>>>(fGates, fH, fC, fHout, hCat, rel, t);
    }

    // ---- heads + decoder --------------------------------------------------
    k_glo  <<<gb(64 * 544), 256, 0, stream>>>(fHout, fGoa, fGlo, hGlo);
    k_heads<<<1, 64, 0, stream>>>(fGlo, Wa, ba, Wv, bv, avail,
                                  out + 524288, out + 524800);
    // s1 = leaky_relu(glo @ Wt1, 0.1)   [64,544] x [544,4096] -> [64,256,4,4]
    gemm(hGlo, hWt1T, fBt1e, fS1, nullptr, 64, 4096, 544,
         544, 544, 4096, 0, 0, 0, 0, 0, 0, 2, 1);
    k_conv3   <<<gb(64 * 128 * 16),  256, 0, stream>>>(fS1, Wc2, bc2, fC2, 64, 256, 128, 4, 4);
    k_conv3_up<<<gb(64 * 64 * 64),   256, 0, stream>>>(fC2, Wc3, bc3, fC3, 64, 128, 64, 4, 4);
    k_conv3_up<<<gb(64 * 64 * 256),  256, 0, stream>>>(fC3, Wc4, bc4, fC4, 64, 64, 64, 8, 8);
    k_conv3_up<<<gb(64 * 64 * 1024), 256, 0, stream>>>(fC4, Wc5, bc5, fC5, 64, 64, 64, 16, 16);
    k_conv3_up<<<gb(64 * 2 * 4096),  256, 0, stream>>>(fC5, Wc6, bc6, fC6, 64, 64, 2, 32, 32);
    k_softmax_sp<<<128, 256, 0, stream>>>(fC6, out);
    k_outlstm<<<gb(32768), 256, 0, stream>>>(fH, fC, out + 524864);
}
